// Net_55714315763758
// MI455X (gfx1250) — matrix operand order repaired and checked by reading, not compiled, not run on hardware
//
#include <hip/hip_runtime.h>

// ---------------- problem constants (from reference) ----------------
#define N_NODES 40000
#define N_EDGES 640000
#define IN_CH   128
#define HID_CH  128
#define OUT_CH  64

typedef __attribute__((ext_vector_type(16))) __bf16 v16bf;
typedef __attribute__((ext_vector_type(8)))  float  v8f;

// ---------------------------------------------------------------
// Edge scatter-add: agg[dst, :] += x[src, :]
// One thread handles 4 channels (float4 read, 4 f32 L2 atomics).
// ---------------------------------------------------------------
__global__ void scatter_add_kernel(const float* __restrict__ x,
                                   const int*   __restrict__ ei,
                                   float*       __restrict__ agg,
                                   int n_edges, int ch)
{
    const int lanes_per_edge = ch >> 2;            // 32 for ch=128, 16 for ch=64
    long long idx = (long long)blockIdx.x * blockDim.x + threadIdx.x;
    int e = (int)(idx / lanes_per_edge);
    int c = ((int)(idx % lanes_per_edge)) << 2;
    if (e >= n_edges) return;

    int src = ei[e];
    int dst = ei[n_edges + e];

    const float4 v = *(const float4*)(x + (long long)src * ch + c);
    float* o = agg + (long long)dst * ch + c;
    atomicAdd(o + 0, v.x);
    atomicAdd(o + 1, v.y);
    atomicAdd(o + 2, v.z);
    atomicAdd(o + 3, v.w);
}

// ---------------------------------------------------------------
// Register-blocked strip GEMM: one wave32 computes a 16-node x
// (OTW*16)-outch strip, so A fragments are loaded/converted ONCE per
// k-step and reused across OTW (x2 if DUAL) WMMAs.
//   out = [relu]( [A1 @ W1^T] + A2 @ W2^T + [bias] + [addc] )
//   A1,A2 : [N_NODES, 128] f32     W1,W2 : [OUTC, 128] f32 row-major
//   addc  : [N_NODES, OUTC] f32 elementwise term preloaded into C
//
// VGPR layouts per CDNA5 ISA (CDNA5 ISA):
//   A 16x32 bf16: lane m=lane%16, half=lane/16.
//     elem j: K = (j<8 ? 0 : 16) + half*8 + (j%8)  within k-tile
//     -> two contiguous 8-float runs at [half*8, +8) and [16+half*8, +8)
//   B 32x16 bf16: lane n=lane%16, half=lane/16.
//     elem j: K = (j<8 ? 0 : 16) + half*8 + (j%8)  (same as A)
//   C/D 16x16 f32: vgpr r -> M = r + half*8, N = lane%16
// ---------------------------------------------------------------
template <int OUTC, int OTW, bool DUAL, bool ADDC, bool BIASF, bool RELU>
__global__ void gemm_strip_kernel(const float* __restrict__ A1,
                                  const float* __restrict__ A2,
                                  const float* __restrict__ W1,
                                  const float* __restrict__ W2,
                                  const float* __restrict__ bias,
                                  const float* __restrict__ addc,
                                  float*       __restrict__ out)
{
    constexpr int K      = 128;
    constexpr int OTILES = OUTC / 16;
    constexpr int GROUPS = OTILES / OTW;
    constexpr int NTILES = N_NODES / 16;

    const int wave = blockIdx.x * (blockDim.x >> 5) + (threadIdx.x >> 5);
    if (wave >= NTILES * GROUPS) return;           // wave-uniform: EXEC stays all-1s

    const int ntile = wave / GROUPS;
    const int og    = (wave % GROUPS) * OTW;       // first otile of this strip
    const int lane  = threadIdx.x & 31;
    const int mh    = lane & 15;
    const int half  = lane >> 4;

    const float* __restrict__ a2p = A2 + (long long)(ntile * 16 + mh) * K + half * 8;
    const float* __restrict__ a1p = A1 + (long long)(ntile * 16 + mh) * K + half * 8;

    v8f acc[OTW];
#pragma unroll
    for (int t = 0; t < OTW; ++t) {
        const float bv = BIASF ? bias[(og + t) * 16 + mh] : 0.0f;
#pragma unroll
        for (int r = 0; r < 8; ++r) acc[t][r] = bv;
        if (ADDC) {
            const float* __restrict__ cp =
                addc + (long long)(ntile * 16) * OUTC + (og + t) * 16 + mh;
#pragma unroll
            for (int r = 0; r < 8; ++r)
                acc[t][r] += cp[(long long)(r + half * 8) * OUTC];
        }
    }

#pragma unroll
    for (int kt = 0; kt < K; kt += 32) {
        // ---- load + convert A fragments once per k-step ----
        v16bf a2, a1;
#pragma unroll
        for (int j = 0; j < 8; ++j) {
            a2[j]     = (__bf16)a2p[kt + j];
            a2[j + 8] = (__bf16)a2p[kt + 16 + j];
        }
        if (DUAL) {
#pragma unroll
            for (int j = 0; j < 8; ++j) {
                a1[j]     = (__bf16)a1p[kt + j];
                a1[j + 8] = (__bf16)a1p[kt + 16 + j];
            }
        }
        // ---- reuse A across the whole output strip ----
#pragma unroll
        for (int t = 0; t < OTW; ++t) {
            if (DUAL) {
                const float* __restrict__ b1p =
                    W1 + (long long)((og + t) * 16 + mh) * K + half * 8 + kt;
                v16bf b1;
#pragma unroll
                for (int j = 0; j < 8; ++j) { b1[j] = (__bf16)b1p[j]; b1[j + 8] = (__bf16)b1p[16 + j]; }
                acc[t] = __builtin_amdgcn_wmma_f32_16x16x32_bf16(
                             false, a1, false, b1, (short)0, acc[t], false, false);
            }
            const float* __restrict__ b2p =
                W2 + (long long)((og + t) * 16 + mh) * K + half * 8 + kt;
            v16bf b2;
#pragma unroll
            for (int j = 0; j < 8; ++j) { b2[j] = (__bf16)b2p[j]; b2[j + 8] = (__bf16)b2p[16 + j]; }
            acc[t] = __builtin_amdgcn_wmma_f32_16x16x32_bf16(
                         false, a2, false, b2, (short)0, acc[t], false, false);
        }
    }

    // ---- store D: row = ntile*16 + r + half*8, col = (og+t)*16 + mh ----
#pragma unroll
    for (int t = 0; t < OTW; ++t) {
        float* __restrict__ op =
            out + (long long)(ntile * 16) * OUTC + (og + t) * 16 + mh;
#pragma unroll
        for (int r = 0; r < 8; ++r) {
            float v = acc[t][r];
            if (RELU) v = fmaxf(v, 0.0f);
            op[(long long)(r + half * 8) * OUTC] = v;
        }
    }
}

// ---------------------------------------------------------------
// Pipeline (exploiting linearity of segment_sum for layer 2):
//   1. memset agg1[N,128]; scatter x (128 ch)
//   2. h = relu(agg1 @ W1_rel^T + x @ W1_root^T + b1)
//   3. z = h @ W2_rel^T                       [N,64]  (project BEFORE scatter)
//   4. memset agg2[N,64]; scatter z (64 ch)   -> half the atomics of 128-ch
//   5. out = agg2 + h @ W2_root^T + b2
// ---------------------------------------------------------------
extern "C" void kernel_launch(void* const* d_in, const int* in_sizes, int n_in,
                              void* d_out, int out_size, void* d_ws, size_t ws_size,
                              hipStream_t stream)
{
    const float* x       = (const float*)d_in[0];
    const int*   ei      = (const int*)  d_in[1];
    const float* W1_rel  = (const float*)d_in[2];
    const float* W1_root = (const float*)d_in[3];
    const float* b1      = (const float*)d_in[4];
    const float* W2_rel  = (const float*)d_in[5];
    const float* W2_root = (const float*)d_in[6];
    const float* b2      = (const float*)d_in[7];
    float*       out     = (float*)d_out;

    // ws layout: [ agg1 : N*128 ][ h : N*128 ]
    // After layer 1, agg1 region is dead -> reuse it for z [N,64] + agg2 [N,64].
    float* agg1 = (float*)d_ws;
    float* h    = agg1 + (size_t)N_NODES * HID_CH;
    float* z    = agg1;                                   // [N, 64]
    float* agg2 = agg1 + (size_t)N_NODES * OUT_CH;        // [N, 64]

    const int blk = 256;                                  // 8 waves / block

    // ---- layer 1: scatter x, dual GEMM with relu ----
    hipMemsetAsync(agg1, 0, (size_t)N_NODES * HID_CH * sizeof(float), stream);
    {
        long long threads = (long long)N_EDGES * (IN_CH / 4);
        int grid = (int)((threads + blk - 1) / blk);
        scatter_add_kernel<<<grid, blk, 0, stream>>>(x, ei, agg1, N_EDGES, IN_CH);
    }
    {
        int waves = (N_NODES / 16) * ((HID_CH / 16) / 4); // 5000 strips of 4 otiles
        gemm_strip_kernel<HID_CH, 4, true, false, true, true>
            <<<(waves + 7) / 8, blk, 0, stream>>>(agg1, x, W1_rel, W1_root, b1,
                                                  nullptr, h);
    }

    // ---- layer 2a: z = h @ W2_rel^T (project to 64 ch before scatter) ----
    {
        int waves = (N_NODES / 16) * ((OUT_CH / 16) / 4); // 2500 strips
        gemm_strip_kernel<OUT_CH, 4, false, false, false, false>
            <<<(waves + 7) / 8, blk, 0, stream>>>(nullptr, h, nullptr, W2_rel,
                                                  nullptr, nullptr, z);
    }

    // ---- layer 2b: scatter z at 64 channels (half the atomic traffic) ----
    hipMemsetAsync(agg2, 0, (size_t)N_NODES * OUT_CH * sizeof(float), stream);
    {
        long long threads = (long long)N_EDGES * (OUT_CH / 4);
        int grid = (int)((threads + blk - 1) / blk);
        scatter_add_kernel<<<grid, blk, 0, stream>>>(z, ei, agg2, N_EDGES, OUT_CH);
    }

    // ---- layer 2c: out = agg2 + h @ W2_root^T + b2 ----
    {
        int waves = (N_NODES / 16) * ((OUT_CH / 16) / 4); // 2500 strips
        gemm_strip_kernel<OUT_CH, 4, false, true, true, false>
            <<<(waves + 7) / 8, blk, 0, stream>>>(nullptr, h, nullptr, W2_root,
                                                  b2, agg2, out);
    }
}
